// GroupedQueryAttention_46591805227232
// MI455X (gfx1250) — matrix operand order repaired and checked by reading, not compiled, not run on hardware
//
#include <hip/hip_runtime.h>

#define DIM     2048
#define HQ      16
#define HKV     4
#define HD      128
#define KV_DIM  512
#define BATCH   2
#define SEQ     2048
#define THETA   10000.0f
#define ROPE_F  8.0f
#define MAXLEN  4096
#define SCALE_Q 0.08838834764831845f   // 128^-0.5

typedef __attribute__((ext_vector_type(16))) __bf16 v16bf;
typedef __attribute__((ext_vector_type(8)))  float  v8f;

struct U4 { unsigned int x[4]; };
union alignas(32) FragBF { v16bf v; U4 u[2]; };

static __device__ __forceinline__ unsigned short f32_to_bf16(float f) {
    union { float f; unsigned int u; } x; x.f = f;
    unsigned int u = x.u;
    u += ((u >> 16) & 1u) + 0x7FFFu;      // round-to-nearest-even
    return (unsigned short)(u >> 16);
}
static __device__ __forceinline__ float bf16_to_f32(unsigned short h) {
    union { unsigned int u; float f; } x; x.u = ((unsigned int)h) << 16;
    return x.f;
}
static __device__ __forceinline__ U4 ldg4(const unsigned short* p) {
    return *reinterpret_cast<const U4*>(p);
}

// ---- async global->LDS staging (gfx1250 GLOBAL_LOAD_ASYNC_TO_LDS_B128 path) ----
#define HAS_ASYNC_LDS __has_builtin(__builtin_amdgcn_global_load_async_to_lds_b128)

#if HAS_ASYNC_LDS
typedef int v4i_ __attribute__((vector_size(16)));
typedef __attribute__((address_space(1))) v4i_ as1_v4i;
typedef __attribute__((address_space(3))) v4i_ as3_v4i;
#endif

static __device__ __forceinline__ void stage16B(const unsigned short* gp, unsigned short* lp) {
#if HAS_ASYNC_LDS
    __builtin_amdgcn_global_load_async_to_lds_b128(
        (as1_v4i*)(unsigned long long)gp,
        (as3_v4i*)(unsigned int)(unsigned long long)lp, 0, 0);
#else
    *reinterpret_cast<U4*>(lp) = *reinterpret_cast<const U4*>(gp);
#endif
}
static __device__ __forceinline__ void stage_wait() {
#if HAS_ASYNC_LDS
#if __has_builtin(__builtin_amdgcn_s_wait_asynccnt)
    __builtin_amdgcn_s_wait_asynccnt(0);
#else
    asm volatile("s_wait_asynccnt 0x0" ::: "memory");
#endif
#endif
}

// ---------------------------------------------------------------- cvt f32->bf16
__global__ void cvt_bf16_kernel(const float* __restrict__ in,
                                unsigned short* __restrict__ out, int n) {
    int i = blockIdx.x * blockDim.x + threadIdx.x;
    if (i < n) out[i] = f32_to_bf16(in[i]);
}

// ---------------------------------------------------------------- GEMM  C = A @ B^T + bias
// A: [M,K] bf16 row-major.  Bm: [N,K] bf16 row-major (i.e. W).
// Register-blocked: one wave computes a 64x64 output tile (4x4 WMMA accumulators).
template <bool BF16OUT>
__global__ __launch_bounds__(256) void gemm_nt_kernel(
    const unsigned short* __restrict__ A,
    const unsigned short* __restrict__ Bm,
    const float* __restrict__ bias,
    void* __restrict__ Cout, int M, int N, int K)
{
    int wave = threadIdx.x >> 5;
    int lane = threadIdx.x & 31;
    int r = lane & 15;          // row-within-tile (A) / col-within-tile (B,C)
    int h = lane >> 4;          // lane half
    int tilesN = N >> 6;        // 64-wide column tiles
    int t  = blockIdx.x * 8 + wave;
    int m0 = (t / tilesN) << 6;
    int n0 = (t % tilesN) << 6;

    const unsigned short* ar[4];
    const unsigned short* br[4];
#pragma unroll
    for (int i = 0; i < 4; ++i) {
        ar[i] = A  + (size_t)(m0 + i * 16 + r) * K;
        br[i] = Bm + (size_t)(n0 + i * 16 + r) * K;
    }

    v8f acc[4][4];
#pragma unroll
    for (int i = 0; i < 4; ++i)
#pragma unroll
        for (int j = 0; j < 4; ++j)
            acc[i][j] = (v8f){0.f,0.f,0.f,0.f,0.f,0.f,0.f,0.f};

    for (int kk = 0; kk < K; kk += 32) {
        FragBF af[4], bf[4];
#pragma unroll
        for (int i = 0; i < 4; ++i) {
            // A layout: e<8 -> k=e+8h ; e>=8 -> k=8+e+8h  (two 8-elem chunks)
            af[i].u[0] = ldg4(ar[i] + kk + 8 * h);
            af[i].u[1] = ldg4(ar[i] + kk + 16 + 8 * h);
            // B layout: e<8 -> k=e+8h ; e>=8 -> k=8+e+8h  (two 8-elem chunks)
            bf[i].u[0] = ldg4(br[i] + kk + 8 * h);
            bf[i].u[1] = ldg4(br[i] + kk + 16 + 8 * h);
        }
        if (kk + 256 < K) {     // pull next k-slab toward the caches (global_prefetch)
#pragma unroll
            for (int i = 0; i < 4; ++i) {
                __builtin_prefetch(ar[i] + kk + 256, 0, 0);
                __builtin_prefetch(br[i] + kk + 256, 0, 0);
            }
        }
#pragma unroll
        for (int i = 0; i < 4; ++i)
#pragma unroll
            for (int j = 0; j < 4; ++j)
                acc[i][j] = __builtin_amdgcn_wmma_f32_16x16x32_bf16(
                    false, af[i].v, false, bf[j].v, (short)0, acc[i][j], false, false);
    }

#pragma unroll
    for (int j = 0; j < 4; ++j) {
        float bv = bias[n0 + j * 16 + r];
#pragma unroll
        for (int i = 0; i < 4; ++i) {
#pragma unroll
            for (int vr = 0; vr < 8; ++vr) {
                int row = m0 + i * 16 + vr + 8 * h;
                float val = acc[i][j][vr] + bv;
                if (BF16OUT)
                    ((unsigned short*)Cout)[(size_t)row * N + n0 + j * 16 + r] = f32_to_bf16(val);
                else
                    ((float*)Cout)[(size_t)row * N + n0 + j * 16 + r] = val;
            }
        }
    }
}

// ---------------------------------------------------------------- RoPE (in-place, bf16)
__global__ void rope_kernel(unsigned short* __restrict__ x, int H, float outscale) {
    int idx = blockIdx.x * blockDim.x + threadIdx.x;
    int total = BATCH * SEQ * H * (HD / 2);
    if (idx >= total) return;
    int i  = idx % (HD / 2);
    int t  = idx / (HD / 2);
    int hh = t % H;   t /= H;
    int n  = t % SEQ;
    int b  = t / SEQ;
    // seq_len_eff = max(SEQ, MAXLEN) = MAXLEN here
    float base_adj = powf(ROPE_F * (float)MAXLEN / (float)MAXLEN - (ROPE_F - 1.0f),
                          (float)HD / (float)(HD - 2));
    float inv = powf(THETA * base_adj, -(float)i / (float)(HD / 2));
    float ang = (float)n * inv;
    float c = cosf(ang), s = sinf(ang);
    size_t base = (((size_t)b * SEQ + n) * H + hh) * HD;
    float x1 = bf16_to_f32(x[base + i]);
    float x2 = bf16_to_f32(x[base + i + HD / 2]);
    x[base + i]          = f32_to_bf16((x1 * c - x2 * s) * outscale);
    x[base + i + HD / 2] = f32_to_bf16((x2 * c + x1 * s) * outscale);
}

// ---------------------------------------------------------------- V transpose -> [B,HKV,HD,SEQ]
__global__ void vtrans_kernel(const unsigned short* __restrict__ vbf,
                              unsigned short* __restrict__ vt) {
    int idx = blockIdx.x * blockDim.x + threadIdx.x;
    int total = BATCH * SEQ * HKV * HD;
    if (idx >= total) return;
    int d = idx % HD;  int t = idx / HD;
    int hh = t % HKV;  t /= HKV;
    int s = t % SEQ;   int b = t / SEQ;
    vt[(((size_t)b * HKV + hh) * HD + d) * SEQ + s] = vbf[idx];
}

// ---------------------------------------------------------------- Flash attention
// grid: (SEQ/64, BATCH*HQ); block 128 (4 waves); wave owns a 16-row query strip.
// K/V tiles are staged into LDS once per block (async global->LDS), shared by all waves.
__global__ __launch_bounds__(128) void attn_kernel(
    const unsigned short* __restrict__ qbf,   // [B,SEQ,HQ,HD]  (pre-scaled by SCALE_Q)
    const unsigned short* __restrict__ kbf,   // [B,SEQ,HKV,HD]
    const unsigned short* __restrict__ vt,    // [B,HKV,HD,SEQ]
    unsigned short* __restrict__ attn)        // [B,SEQ,HQ,HD]
{
    __shared__ alignas(16) unsigned short kt[32 * HD];      // 8KB  [key_local][d]
    __shared__ alignas(16) unsigned short vtile[HD * 32];   // 8KB  [d][key_local]
    __shared__ alignas(16) unsigned short plds[4][16 * 32]; // 4KB  per-wave P transpose

    int wave = threadIdx.x >> 5;
    int lane = threadIdx.x & 31;
    int r = lane & 15;
    int h = lane >> 4;
    int m0blk = blockIdx.x * 64;
    int m0 = m0blk + wave * 16;
    int bh = blockIdx.y;
    int b  = bh / HQ;
    int hq = bh % HQ;
    int hk = hq / (HQ / HKV);

    const unsigned short* qb = qbf + ((size_t)(b * SEQ) * HQ  + hq) * HD;
    const unsigned short* kb = kbf + ((size_t)(b * SEQ) * HKV + hk) * HD;
    const unsigned short* vb = vt  + ((size_t)(b * HKV + hk) * HD) * SEQ;

    // Preload Q fragments: 16 rows x 128 dims = 4 A-fragments
    FragBF qf[4];
    const unsigned short* qrow = qb + (size_t)(m0 + r) * (HQ * HD);
#pragma unroll
    for (int ks = 0; ks < 4; ++ks) {
        qf[ks].u[0] = ldg4(qrow + ks * 32 + 8 * h);
        qf[ks].u[1] = ldg4(qrow + ks * 32 + 16 + 8 * h);
    }

    v8f o[8];
#pragma unroll
    for (int j = 0; j < 8; ++j) o[j] = (v8f){0.f,0.f,0.f,0.f,0.f,0.f,0.f,0.f};
    float mrow[8], lrow[8];
#pragma unroll
    for (int vr = 0; vr < 8; ++vr) { mrow[vr] = -1e30f; lrow[vr] = 0.0f; }

    // Block-uniform key-block count (largest strip in block); extra blocks for lower
    // strips are fully masked and provably no-ops (mrow finite after block 0).
    int nkb = (m0blk + 64 + 31) / 32;
    for (int kb2 = 0; kb2 < nkb; ++kb2) {
        int k0 = kb2 * 32;
        // ---- cooperative async staging: K tile (32x128) and V^T tile (128x32) ----
#pragma unroll
        for (int i = 0; i < 4; ++i) {
            int c = (int)threadIdx.x + 128 * i;              // 512 x 16B chunks
            stage16B(kb + (size_t)(k0 + (c >> 4)) * (HKV * HD) + (c & 15) * 8,
                     kt + (c >> 4) * HD + (c & 15) * 8);
        }
#pragma unroll
        for (int i = 0; i < 4; ++i) {
            int c = (int)threadIdx.x + 128 * i;              // 512 x 16B chunks
            stage16B(vb + (size_t)(c >> 2) * SEQ + k0 + (c & 3) * 8,
                     vtile + (c >> 2) * 32 + (c & 3) * 8);
        }
        stage_wait();
        __syncthreads();

        // ---- scores S = Q @ K^T (fragments from LDS) ----
        v8f s0 = {0.f,0.f,0.f,0.f,0.f,0.f,0.f,0.f};
        v8f s1 = {0.f,0.f,0.f,0.f,0.f,0.f,0.f,0.f};
        const unsigned short* krow0 = kt + r * HD;
        const unsigned short* krow1 = kt + (16 + r) * HD;
#pragma unroll
        for (int ks = 0; ks < 4; ++ks) {
            FragBF f0, f1;
            f0.u[0] = ldg4(krow0 + ks * 32 + 8 * h);
            f0.u[1] = ldg4(krow0 + ks * 32 + 16 + 8 * h);
            f1.u[0] = ldg4(krow1 + ks * 32 + 8 * h);
            f1.u[1] = ldg4(krow1 + ks * 32 + 16 + 8 * h);
            s0 = __builtin_amdgcn_wmma_f32_16x16x32_bf16(false, qf[ks].v, false, f0.v,
                                                         (short)0, s0, false, false);
            s1 = __builtin_amdgcn_wmma_f32_16x16x32_bf16(false, qf[ks].v, false, f1.v,
                                                         (short)0, s1, false, false);
        }
        // causal mask (C layout: row = m0+vr+8h, col = k0 + [16*tile] + r)
#pragma unroll
        for (int vr = 0; vr < 8; ++vr) {
            int row = m0 + vr + 8 * h;
            if (k0 + r      > row) s0[vr] = -1e30f;
            if (k0 + 16 + r > row) s1[vr] = -1e30f;
        }
        // online softmax; reductions stay inside each 16-lane half (matches row layout)
#pragma unroll
        for (int vr = 0; vr < 8; ++vr) {
            float t = fmaxf(s0[vr], s1[vr]);
            t = fmaxf(t, __shfl_xor(t, 1, 32));
            t = fmaxf(t, __shfl_xor(t, 2, 32));
            t = fmaxf(t, __shfl_xor(t, 4, 32));
            t = fmaxf(t, __shfl_xor(t, 8, 32));
            float mnew = fmaxf(mrow[vr], t);
            float sc = __expf(mrow[vr] - mnew);
            float p0 = __expf(s0[vr] - mnew);
            float p1 = __expf(s1[vr] - mnew);
            s0[vr] = p0; s1[vr] = p1;
            float rs = p0 + p1;
            rs += __shfl_xor(rs, 1, 32);
            rs += __shfl_xor(rs, 2, 32);
            rs += __shfl_xor(rs, 4, 32);
            rs += __shfl_xor(rs, 8, 32);
            lrow[vr] = lrow[vr] * sc + rs;
            mrow[vr] = mnew;
#pragma unroll
            for (int j = 0; j < 8; ++j) o[j][vr] *= sc;
        }
        // Transpose P (16x32) C-layout -> A-layout via per-wave LDS
        volatile unsigned short* pl = plds[wave];
#pragma unroll
        for (int vr = 0; vr < 8; ++vr) {
            int rowl = vr + 8 * h;
            pl[rowl * 32 + r]      = f32_to_bf16(s0[vr]);
            pl[rowl * 32 + 16 + r] = f32_to_bf16(s1[vr]);
        }
        asm volatile("" ::: "memory");
        __builtin_amdgcn_wave_barrier();
        FragBF pf;
        {
            const unsigned short* base = (const unsigned short*)plds[wave] + r * 32;
            pf.u[0] = *reinterpret_cast<const U4*>(base + 8 * h);
            pf.u[1] = *reinterpret_cast<const U4*>(base + 16 + 8 * h);
        }
        asm volatile("" ::: "memory");
        __builtin_amdgcn_wave_barrier();
        // O += P @ V   (V^T fragments from LDS)
#pragma unroll
        for (int j = 0; j < 8; ++j) {
            const unsigned short* vrow = vtile + (j * 16 + r) * 32;
            FragBF vf;
            vf.u[0] = ldg4(vrow + 8 * h);
            vf.u[1] = ldg4(vrow + 16 + 8 * h);
            o[j] = __builtin_amdgcn_wmma_f32_16x16x32_bf16(false, pf.v, false, vf.v,
                                                           (short)0, o[j], false, false);
        }
        __syncthreads();   // protect kt/vtile before next iteration's staging
    }

    unsigned short* ob = attn + ((size_t)(b * SEQ) * HQ + hq) * HD;
#pragma unroll
    for (int vr = 0; vr < 8; ++vr) {
        float inv = 1.0f / fmaxf(lrow[vr], 1e-30f);
        int row = m0 + vr + 8 * h;
#pragma unroll
        for (int j = 0; j < 8; ++j)
            ob[(size_t)row * (HQ * HD) + j * 16 + r] = f32_to_bf16(o[j][vr] * inv);
    }
}

// ---------------------------------------------------------------- host launch
extern "C" void kernel_launch(void* const* d_in, const int* in_sizes, int n_in,
                              void* d_out, int out_size, void* d_ws, size_t ws_size,
                              hipStream_t stream) {
    (void)in_sizes; (void)n_in; (void)out_size; (void)ws_size;
    const float* query = (const float*)d_in[0];
    const float* key   = (const float*)d_in[1];
    const float* value = (const float*)d_in[2];
    const float* Wq    = (const float*)d_in[3];
    const float* bq    = (const float*)d_in[4];
    const float* Wk    = (const float*)d_in[5];
    const float* bk    = (const float*)d_in[6];
    const float* Wv    = (const float*)d_in[7];
    const float* bv    = (const float*)d_in[8];
    const float* Wo    = (const float*)d_in[9];
    const float* bo    = (const float*)d_in[10];
    float* out = (float*)d_out;

    char* ws = (char*)d_ws;
    unsigned short* qbf    = (unsigned short*)ws; ws += (size_t)BATCH * SEQ * DIM    * 2; // 16MB
    unsigned short* kbf    = (unsigned short*)ws; ws += (size_t)BATCH * SEQ * KV_DIM * 2; //  4MB
    unsigned short* vbf    = (unsigned short*)ws; ws += (size_t)BATCH * SEQ * KV_DIM * 2; //  4MB
    unsigned short* vtb    = (unsigned short*)ws; ws += (size_t)BATCH * SEQ * KV_DIM * 2; //  4MB
    unsigned short* attnb  = (unsigned short*)ws; ws += (size_t)BATCH * SEQ * DIM    * 2; // 16MB
    unsigned short* xstage = (unsigned short*)ws; ws += (size_t)BATCH * SEQ * DIM    * 2; // 16MB
    unsigned short* wstage = (unsigned short*)ws;                                        //  8MB

    const int M = BATCH * SEQ;                   // 4096
    auto cvt = [&](const float* in, unsigned short* o2, int n) {
        cvt_bf16_kernel<<<(n + 255) / 256, 256, 0, stream>>>(in, o2, n);
    };
    auto gemm_blocks = [](int m, int n) { return ((m / 64) * (n / 64)) / 8; };

    // Q = X @ Wq^T + bq
    cvt(query, xstage, M * DIM);
    cvt(Wq, wstage, DIM * DIM);
    gemm_nt_kernel<true><<<gemm_blocks(M, DIM), 256, 0, stream>>>(
        xstage, wstage, bq, qbf, M, DIM, DIM);
    // K = X @ Wk^T + bk
    cvt(key, xstage, M * DIM);
    cvt(Wk, wstage, KV_DIM * DIM);
    gemm_nt_kernel<true><<<gemm_blocks(M, KV_DIM), 256, 0, stream>>>(
        xstage, wstage, bk, kbf, M, KV_DIM, DIM);
    // V = X @ Wv^T + bv
    cvt(value, xstage, M * DIM);
    cvt(Wv, wstage, KV_DIM * DIM);
    gemm_nt_kernel<true><<<gemm_blocks(M, KV_DIM), 256, 0, stream>>>(
        xstage, wstage, bv, vbf, M, KV_DIM, DIM);
    // RoPE (Q gets the 1/sqrt(HD) score scale folded in)
    rope_kernel<<<(BATCH * SEQ * HQ  * (HD / 2) + 255) / 256, 256, 0, stream>>>(qbf, HQ,  SCALE_Q);
    rope_kernel<<<(BATCH * SEQ * HKV * (HD / 2) + 255) / 256, 256, 0, stream>>>(kbf, HKV, 1.0f);
    // V transpose -> [B,HKV,HD,SEQ]
    vtrans_kernel<<<(BATCH * SEQ * KV_DIM + 255) / 256, 256, 0, stream>>>(vbf, vtb);
    // Flash attention
    attn_kernel<<<dim3(SEQ / 64, BATCH * HQ), 128, 0, stream>>>(qbf, kbf, vtb, attnb);
    // out = attn @ Wo^T + bo   (fp32 output)
    cvt(Wo, wstage, DIM * DIM);
    gemm_nt_kernel<false><<<gemm_blocks(M, DIM), 256, 0, stream>>>(
        attnb, wstage, bo, out, M, DIM, DIM);
}
